// JKNet_9723805958220
// MI455X (gfx1250) — matrix operand order repaired and checked by reading, not compiled, not run on hardware
//
#include <hip/hip_runtime.h>

static constexpr int NN  = 50000;   // nodes
static constexpr int NE  = 800000;  // edges
static constexpr int NF  = 512;     // input features
static constexpr int NH  = 256;     // hidden
static constexpr int NC  = 40;      // classes
static constexpr int NCP = 48;      // classes padded to 16-col tiles

typedef __attribute__((ext_vector_type(16))) __bf16 v16bf;
typedef __attribute__((ext_vector_type(8)))  float  v8f;

__device__ __forceinline__ unsigned short f2bf(float f) {
  unsigned u = __float_as_uint(f);
  u += 0x7FFFu + ((u >> 16) & 1u);      // round-to-nearest-even
  return (unsigned short)(u >> 16);
}

// ---------------- small elementwise kernels ----------------

__global__ void k_zero_f32(float* __restrict__ p, int n) {
  int i = blockIdx.x * blockDim.x + threadIdx.x;
  if (i < n) p[i] = 0.f;
}

__global__ void k_degree(const int* __restrict__ dst, float* __restrict__ deg, int e) {
  int i = blockIdx.x * blockDim.x + threadIdx.x;
  if (i < e) atomicAdd(&deg[dst[i]], 1.0f);
}

__global__ void k_dinv(float* __restrict__ d, int n) {
  int i = blockIdx.x * blockDim.x + threadIdx.x;
  if (i < n) d[i] = rsqrtf(d[i] + 1.0f);   // deg = 1 + in-degree
}

__global__ void k_f32_to_bf16(const float* __restrict__ in, unsigned short* __restrict__ out, int n) {
  int i = blockIdx.x * blockDim.x + threadIdx.x;
  if (i < n) out[i] = f2bf(in[i]);
}

// transpose-convert weight: in [R x C] f32 (row-major) -> out [C x R] bf16 (row-major)
__global__ void k_cvt_w_T(const float* __restrict__ in, unsigned short* __restrict__ out,
                          int R, int C) {
  int i = blockIdx.x * blockDim.x + threadIdx.x;
  if (i >= R * C) return;
  int r = i / C, c = i - r * C;
  out[(size_t)c * R + r] = f2bf(in[i]);
}

// out_W slice [rowbase..rowbase+255, 0..39] -> transposed bf16 [48 x 256], rows 40..47 zero
__global__ void k_cvt_outw_T(const float* __restrict__ w, unsigned short* __restrict__ out,
                             int rowbase) {
  int i = blockIdx.x * blockDim.x + threadIdx.x;   // over NH*NCP
  if (i >= NH * NCP) return;
  int r = i / NCP, c = i - r * NCP;                // r in [0,256), c in [0,48)
  out[(size_t)c * NH + r] = (c < NC) ? f2bf(w[(size_t)(rowbase + r) * NC + c])
                                     : (unsigned short)0;
}

// ---------------- WMMA bf16 GEMM: C[M,N] = A[M,K] * Bt[N,K]^T (+bias)(+=C) ----------------
// B is supplied TRANSPOSED (N x K row-major) so both fragment loads are contiguous b128s.
// One wave per block; wave computes MT x NT tiles of 16x16.
// Fragment layouts per CDNA5 ISA 7.12.2 (wave32).

template<int MT, int NT>
__global__ __launch_bounds__(32)
void k_gemm_bf16(const unsigned short* __restrict__ A,
                 const unsigned short* __restrict__ Bt,
                 float* __restrict__ C,
                 const float* __restrict__ bias,
                 int M, int K, int lda, int ldbt, int ldc, int ncol, int accum)
{
  const int lane = threadIdx.x;
  const int half = lane >> 4;     // lane half selects K sub-blocks
  const int l16  = lane & 15;
  const int tm0  = blockIdx.y * MT;   // in 16-row tiles
  const int tn0  = blockIdx.x * NT;   // in 16-col tiles

  const v8f vzero = {0.f, 0.f, 0.f, 0.f, 0.f, 0.f, 0.f, 0.f};
  v8f acc[MT][NT];
#pragma unroll
  for (int mi = 0; mi < MT; ++mi)
#pragma unroll
    for (int ni = 0; ni < NT; ++ni)
      acc[mi][ni] = vzero;

  union Frag { v16bf v; unsigned u[8]; };

  for (int k0 = 0; k0 < K; k0 += 32) {
    // A fragments: 16x32 bf16. lane half h, M=l16:
    //   VGPR v<4 -> K = 2v + 8h ; VGPR v>=4 -> K = 16 + 2(v-4) + 8h. Pairs (K,K+1) per reg.
    Frag a[MT];
#pragma unroll
    for (int mi = 0; mi < MT; ++mi) {
      int row = (tm0 + mi) * 16 + l16;
      row = row < M ? row : M - 1;               // clamp; store is guarded
      const unsigned short* ap = A + (size_t)row * lda + k0 + half * 8;
#pragma unroll
      for (int v = 0; v < 8; ++v) {
        const int kb = (v < 4) ? (2 * v) : (8 + 2 * v);  // 16 + 2*(v-4)
        a[mi].u[v] = *(const unsigned*)(ap + kb);
      }
    }
    // B fragments from transposed weights: lane half h, N=l16:
    //   VGPR v<4 -> K = 2v + 8h ; VGPR v>=4 -> K = 16 + 2(v-4) + 8h. Pairs (K,K+1) per reg.
    Frag b[NT];
#pragma unroll
    for (int ni = 0; ni < NT; ++ni) {
      int col = (tn0 + ni) * 16 + l16;
      const unsigned short* bp = Bt + (size_t)col * ldbt + k0 + half * 8;
#pragma unroll
      for (int v = 0; v < 8; ++v)
        b[ni].u[v] = *(const unsigned*)(bp + ((v < 4) ? (2 * v) : (8 + 2 * v)));
    }
#pragma unroll
    for (int mi = 0; mi < MT; ++mi)
#pragma unroll
      for (int ni = 0; ni < NT; ++ni)
        acc[mi][ni] = __builtin_amdgcn_wmma_f32_16x16x32_bf16(
            false, a[mi].v, false, b[ni].v, (short)0, acc[mi][ni], false, false);
  }

  // C/D layout: VGPR r -> M = r + 8*half, N = l16.
#pragma unroll
  for (int ni = 0; ni < NT; ++ni) {
    int col = (tn0 + ni) * 16 + l16;
    if (col >= ncol) continue;
    float bv = bias ? bias[col] : 0.f;
#pragma unroll
    for (int mi = 0; mi < MT; ++mi) {
#pragma unroll
      for (int r = 0; r < 8; ++r) {
        int row = (tm0 + mi) * 16 + half * 8 + r;
        if (row < M) {
          size_t idx = (size_t)row * ldc + col;
          float v = acc[mi][ni][r] + bv;
          if (accum) v += C[idx];
          C[idx] = v;
        }
      }
    }
  }
}

// ---------------- edge aggregation: agg[dst] += hw[src] * dinv[src]*dinv[dst] ----------------
// 64 threads per edge, 4 floats each -> coalesced 1KB rows; atomics resolve at L2
// (hw + agg together = 102 MB, resident in the 192 MB L2).

__global__ void k_scatter(const int* __restrict__ src, const int* __restrict__ dst,
                          const float* __restrict__ dinv,
                          const float* __restrict__ hw, float* __restrict__ agg, int e)
{
  long long t = (long long)blockIdx.x * blockDim.x + threadIdx.x;
  int ei = (int)(t >> 6);
  if (ei >= e) return;
  int c = ((int)t & 63) << 2;
  int s = src[ei], d = dst[ei];
  float nrm = dinv[s] * dinv[d];
  const float4 v = *(const float4*)(hw + (size_t)s * NH + c);
  float* o = agg + (size_t)d * NH + c;
  atomicAdd(o + 0, v.x * nrm);
  atomicAdd(o + 1, v.y * nrm);
  atomicAdd(o + 2, v.z * nrm);
  atomicAdd(o + 3, v.w * nrm);
}

// h = relu(agg + dinv^2 * hw + gc_b + lin_b)  -> bf16 (input of next GEMMs)
__global__ void k_finalize(const float* __restrict__ agg, const float* __restrict__ hw,
                           const float* __restrict__ dinv,
                           const float* __restrict__ b_gc, const float* __restrict__ b_lin,
                           unsigned short* __restrict__ hbf)
{
  int i = blockIdx.x * blockDim.x + threadIdx.x;
  if (i >= NN * NH) return;
  int node = i >> 8;          // NH == 256
  int c = i & (NH - 1);
  float dn = dinv[node];
  float v = agg[i] + dn * dn * hw[i] + b_gc[c] + b_lin[c];
  hbf[i] = f2bf(fmaxf(v, 0.f));
}

// ---------------- launch ----------------

extern "C" void kernel_launch(void* const* d_in, const int* in_sizes, int n_in,
                              void* d_out, int out_size, void* d_ws, size_t ws_size,
                              hipStream_t stream)
{
  const float* x        = (const float*)d_in[0];
  const int*   edges    = (const int*)d_in[1];
  const float* in_gc_W  = (const float*)d_in[2];
  const float* in_gc_b  = (const float*)d_in[3];
  const float* in_lin_W = (const float*)d_in[4];
  const float* in_lin_b = (const float*)d_in[5];
  const float* convs_W  = (const float*)d_in[6];
  const float* convs_b  = (const float*)d_in[7];
  const float* lins_W   = (const float*)d_in[8];
  const float* lins_b   = (const float*)d_in[9];
  const float* out_W    = (const float*)d_in[10];
  const float* out_b    = (const float*)d_in[11];
  float* out = (float*)d_out;

  const int* src = edges;        // edge_index[0]
  const int* dst = edges + NE;   // edge_index[1]

  // workspace layout (256B aligned)
  char* ws = (char*)d_ws;
  size_t off = 0;
  float* dinv = (float*)(ws + off);          off += ((size_t)NN * 4 + 255) / 256 * 256;
  unsigned short* hbf = (unsigned short*)(ws + off); off += (size_t)NN * NF * 2;
  unsigned short* wbf = (unsigned short*)(ws + off); off += (size_t)NF * NH * 2;
  float* hw  = (float*)(ws + off);           off += (size_t)NN * NH * 4;
  float* agg = (float*)(ws + off);           off += (size_t)NN * NH * 4;
  (void)ws_size; (void)in_sizes; (void)n_in; (void)out_size;

  auto cdiv = [](long long a, long long b) { return (int)((a + b - 1) / b); };

  // degree -> dinv = rsqrt(1 + in_degree)
  k_zero_f32<<<cdiv(NN, 256), 256, 0, stream>>>(dinv, NN);
  k_degree<<<cdiv(NE, 256), 256, 0, stream>>>(dst, dinv, NE);
  k_dinv<<<cdiv(NN, 256), 256, 0, stream>>>(dinv, NN);

  const dim3 gemmGridMain(NH / 32, cdiv(NN, 64));  // NT=2 (32 cols), MT=4 (64 rows)
  const dim3 gemmGridOut(1, cdiv(NN, 64));         // NT=3 covers 48 padded cols

  // ---- layer 1 (K = 512) ----
  k_f32_to_bf16<<<cdiv((long long)NN * NF, 256), 256, 0, stream>>>(x, hbf, NN * NF);
  k_cvt_w_T<<<cdiv(NF * NH, 256), 256, 0, stream>>>(in_gc_W, wbf, NF, NH);
  k_gemm_bf16<4, 2><<<gemmGridMain, 32, 0, stream>>>(hbf, wbf, hw, nullptr, NN, NF, NF, NF, NH, NH, 0);
  k_cvt_w_T<<<cdiv(NF * NH, 256), 256, 0, stream>>>(in_lin_W, wbf, NF, NH);
  k_gemm_bf16<4, 2><<<gemmGridMain, 32, 0, stream>>>(hbf, wbf, agg, nullptr, NN, NF, NF, NF, NH, NH, 0);
  k_scatter<<<cdiv((long long)NE * 64, 256), 256, 0, stream>>>(src, dst, dinv, hw, agg, NE);
  k_finalize<<<cdiv((long long)NN * NH, 256), 256, 0, stream>>>(agg, hw, dinv, in_gc_b, in_lin_b, hbf);
  // out contribution of h1 (initializes d_out, adds out_b)
  k_cvt_outw_T<<<cdiv(NH * NCP, 256), 256, 0, stream>>>(out_W, wbf, 0);
  k_gemm_bf16<4, 3><<<gemmGridOut, 32, 0, stream>>>(hbf, wbf, out, out_b, NN, NH, NH, NH, NC, NC, 0);

  // ---- layers 2..4 (K = 256) ----
  for (int i = 0; i < 3; ++i) {
    k_cvt_w_T<<<cdiv(NH * NH, 256), 256, 0, stream>>>(convs_W + (size_t)i * NH * NH, wbf, NH, NH);
    k_gemm_bf16<4, 2><<<gemmGridMain, 32, 0, stream>>>(hbf, wbf, hw, nullptr, NN, NH, NH, NH, NH, NH, 0);
    k_cvt_w_T<<<cdiv(NH * NH, 256), 256, 0, stream>>>(lins_W + (size_t)i * NH * NH, wbf, NH, NH);
    k_gemm_bf16<4, 2><<<gemmGridMain, 32, 0, stream>>>(hbf, wbf, agg, nullptr, NN, NH, NH, NH, NH, NH, 0);
    k_scatter<<<cdiv((long long)NE * 64, 256), 256, 0, stream>>>(src, dst, dinv, hw, agg, NE);
    k_finalize<<<cdiv((long long)NN * NH, 256), 256, 0, stream>>>(agg, hw, dinv,
                                                                  convs_b + (size_t)i * NH,
                                                                  lins_b + (size_t)i * NH, hbf);
    k_cvt_outw_T<<<cdiv(NH * NCP, 256), 256, 0, stream>>>(out_W, wbf, (i + 1) * NH);
    k_gemm_bf16<4, 3><<<gemmGridOut, 32, 0, stream>>>(hbf, wbf, out, nullptr, NN, NH, NH, NH, NC, NC, 1);
  }
}
